// Model_80607946211762
// MI455X (gfx1250) — compile-verified, matrix operand order checked by reading, not run on hardware
//
#include <hip/hip_runtime.h>
#include <hip/hip_bf16.h>

// ---------------------------------------------------------------------------
// Problem constants (match reference)
// ---------------------------------------------------------------------------
constexpr int Nn = 40000;   // nodes
constexpr int Ee = 640000;  // edges
constexpr int Gg = 128;     // graphs
constexpr int Dd = 128;     // embedding dim
constexpr int Ll = 5;       // layers
constexpr float BN_EPS = 1e-5f;

// ---------------------------------------------------------------------------
// CDNA5 WMMA types / helpers (bf16 inputs, f32 accumulate)
// ---------------------------------------------------------------------------
typedef __attribute__((ext_vector_type(16))) __bf16 v16bf;
typedef __attribute__((ext_vector_type(8)))  float  v8f;

__device__ __forceinline__ unsigned short f2bf(float f) {
    union { float f; unsigned u; } c; c.f = f;
    unsigned u = c.u;
    u += 0x7FFFu + ((u >> 16) & 1u);   // round-to-nearest-even
    return (unsigned short)(u >> 16);
}

__device__ __forceinline__ v8f wmma_bf16(v16bf a, v16bf b, v8f c) {
    return __builtin_amdgcn_wmma_f32_16x16x32_bf16(
        /*neg_a=*/false, a, /*neg_b=*/false, b,
        /*c_mod=*/(short)0, c, /*reuse_a=*/false, /*reuse_b=*/false);
}

// Fragment-order addressing (ISA 7.12.2, 16-bit A/B layout):
// element with local k (0..31) lives in lane-half = (k>>3)&1 and
// per-lane position pos = 8*((k>>4)&1) + (k&7)  -> 16 bf16 = 32 B per lane.

// Load a pre-packed fragment: Wf layout [kc][nt][lane][16] u16, contiguous.
__device__ __forceinline__ v16bf load_frag_packed(const unsigned short* __restrict__ Wf,
                                                  int kc, int nt, int lane) {
    const uint4* p = (const uint4*)(Wf + ((((kc << 3) + nt) << 5) + lane) * 16);
    union { v16bf v; uint4 q[2]; } u;
    u.q[0] = p[0];
    u.q[1] = p[1];
    return u.v;
}

// Load a packed fragment from LDS ([m_t][kc][lane][16] u16) -> 2x ds_load_b128
__device__ __forceinline__ v16bf load_frag_lds(const unsigned short* sU,
                                               int m_t, int kc, int lane) {
    const uint4* p = (const uint4*)(sU + ((((m_t << 2) + kc) << 5) + lane) * 16);
    union { v16bf v; uint4 q[2]; } u;
    u.q[0] = p[0];
    u.q[1] = p[1];
    return u.v;
}

// A fragment straight from global f32: row-major matrix `a` (+ optional `b`)
// with row stride `stride` floats. Per-lane data = floats [k0..k0+7] and
// [k0+16..k0+23] of row `row`, where k0 = kc*32 + 8*(lane>>4).
__device__ __forceinline__ v16bf load_frag_A_add(const float* __restrict__ a,
                                                 const float* __restrict__ b,
                                                 size_t rowoff, int k0) {
    union { v16bf v; unsigned short s[16]; } u;
    const float4* pa0 = (const float4*)(a + rowoff + k0);
    const float4* pb0 = (const float4*)(b + rowoff + k0);
    float4 a0 = pa0[0], a1 = pa0[1];
    float4 b0 = pb0[0], b1 = pb0[1];
    u.s[0] = f2bf(a0.x + b0.x); u.s[1] = f2bf(a0.y + b0.y);
    u.s[2] = f2bf(a0.z + b0.z); u.s[3] = f2bf(a0.w + b0.w);
    u.s[4] = f2bf(a1.x + b1.x); u.s[5] = f2bf(a1.y + b1.y);
    u.s[6] = f2bf(a1.z + b1.z); u.s[7] = f2bf(a1.w + b1.w);
    const float4* pa1 = (const float4*)(a + rowoff + k0 + 16);
    const float4* pb1 = (const float4*)(b + rowoff + k0 + 16);
    float4 a2 = pa1[0], a3 = pa1[1];
    float4 b2 = pb1[0], b3 = pb1[1];
    u.s[8]  = f2bf(a2.x + b2.x); u.s[9]  = f2bf(a2.y + b2.y);
    u.s[10] = f2bf(a2.z + b2.z); u.s[11] = f2bf(a2.w + b2.w);
    u.s[12] = f2bf(a3.x + b3.x); u.s[13] = f2bf(a3.y + b3.y);
    u.s[14] = f2bf(a3.z + b3.z); u.s[15] = f2bf(a3.w + b3.w);
    return u.v;
}

__device__ __forceinline__ v16bf load_frag_A_plain(const float* __restrict__ a,
                                                   size_t rowoff, int k0) {
    union { v16bf v; unsigned short s[16]; } u;
    const float4* pa0 = (const float4*)(a + rowoff + k0);
    float4 a0 = pa0[0], a1 = pa0[1];
    u.s[0] = f2bf(a0.x); u.s[1] = f2bf(a0.y);
    u.s[2] = f2bf(a0.z); u.s[3] = f2bf(a0.w);
    u.s[4] = f2bf(a1.x); u.s[5] = f2bf(a1.y);
    u.s[6] = f2bf(a1.z); u.s[7] = f2bf(a1.w);
    const float4* pa1 = (const float4*)(a + rowoff + k0 + 16);
    float4 a2 = pa1[0], a3 = pa1[1];
    u.s[8]  = f2bf(a2.x); u.s[9]  = f2bf(a2.y);
    u.s[10] = f2bf(a2.z); u.s[11] = f2bf(a2.w);
    u.s[12] = f2bf(a3.x); u.s[13] = f2bf(a3.y);
    u.s[14] = f2bf(a3.z); u.s[15] = f2bf(a3.w);
    return u.v;
}

// Fragment-order LDS index for element (m_local, k_global<128) of a 64/128-row
// intermediate stored as [m_t][kc][lane][16] u16.
__device__ __forceinline__ int frag_idx(int m, int k) {
    int m_t = m >> 4;
    int kc  = k >> 5;
    int kl  = k & 31;
    int ln  = (m & 15) + ((kl >> 3) & 1) * 16;
    int ps  = ((kl >> 4) & 1) * 8 + (kl & 7);
    return ((((m_t << 2) + kc) << 5) + ln) * 16 + ps;
}

// ---------------------------------------------------------------------------
// Weight pre-pack: W [128x128] row-major (k x n) f32 -> fragment-order bf16
// Wf layout: [kc(4)][nt(8)][lane(32)][pos(16)] u16  (16384 elements)
// ---------------------------------------------------------------------------
__global__ void pack_weight_kernel(const float* __restrict__ W,
                                   unsigned short* __restrict__ Wf) {
    int i = blockIdx.x * 256 + threadIdx.x;
    if (i >= 128 * 128) return;
    int pos   = i & 15;
    int chunk = i >> 4;
    int lane  = chunk & 31;
    int tile  = chunk >> 5;
    int nt    = tile & 7;
    int kc    = tile >> 3;
    int half  = lane >> 4;
    int n     = nt * 16 + (lane & 15);
    int k     = kc * 32 + ((pos >> 3) << 4) + half * 8 + (pos & 7);
    Wf[i] = f2bf(W[k * 128 + n]);
}

// ---------------------------------------------------------------------------
// Utility kernels
// ---------------------------------------------------------------------------
__global__ void zero_kernel(float* __restrict__ p, int n) {
    int i = blockIdx.x * 256 + threadIdx.x;
    if (i < n) p[i] = 0.f;
}

__global__ void scatter_d1_kernel(const float* __restrict__ x,
                                  const float* __restrict__ ew,
                                  const int* __restrict__ src,
                                  const int* __restrict__ dst,
                                  float* __restrict__ agg0, int E) {
    int e = blockIdx.x * 256 + threadIdx.x;
    if (e >= E) return;
    atomicAdd(&agg0[dst[e]], ew[e] * x[src[e]]);
}

__global__ void scatter_d128_kernel(const float* __restrict__ h,
                                    const float* __restrict__ ew,
                                    const int* __restrict__ src,
                                    const int* __restrict__ dst,
                                    float* __restrict__ agg, int E) {
    int gid = blockIdx.x * 256 + threadIdx.x;
    int e = gid >> 5;
    if (e >= E) return;
    int c = (gid & 31) * 4;
    int s = src[e], d = dst[e];
    float w = ew[e];
    const float4 hv = *(const float4*)(h + (size_t)s * Dd + c);
    float* ap = agg + (size_t)d * Dd + c;
    atomicAdd(ap + 0, hv.x * w);
    atomicAdd(ap + 1, hv.y * w);
    atomicAdd(ap + 2, hv.z * w);
    atomicAdd(ap + 3, hv.w * w);
}

// ---------------------------------------------------------------------------
// Fused GIN MLP:  t = relu((agg+h) @ W1 + b1) @ W2 + b2   (+ BN statistics)
// 64 nodes per block, 8 waves. A fragments loaded straight from global,
// B fragments from pre-packed weights, intermediate in LDS fragment order.
// t_out may alias agg: all agg reads happen before the mid-kernel barrier,
// all t_out writes after it.
// ---------------------------------------------------------------------------
__global__ __launch_bounds__(256) void mlp_kernel(
    const float* __restrict__ agg,          // [N,128] or (layer0) agg0 [N]
    const float* __restrict__ h,            // [N,128] or (layer0) x    [N]
    const float* __restrict__ W1_0,         // layer0: [128] first-linear row
    const unsigned short* __restrict__ W1f, // packed [128x128] (layers >=1)
    const float* __restrict__ b1,           // [128]
    const unsigned short* __restrict__ W2f, // packed [128x128]
    const float* __restrict__ b2,           // [128]
    float* __restrict__ t_out,              // [N,128]
    float* __restrict__ sums,               // [256] : sum | sumsq
    int N, int layer0) {
    __shared__ unsigned short sU[64 * 128];   // 16 KB, fragment order

    const int tid    = threadIdx.x;
    const int row0   = blockIdx.x * 64;
    const int wave   = tid >> 5;
    const int lane   = tid & 31;
    const int m_base = (wave & 3) * 16;       // 4 row tiles, 2 waves each
    const int nt0    = (wave >> 2) * 4;       // each wave: 4 of 8 col tiles
    const int half   = (lane >> 4) & 1;
    const int ncol   = lane & 15;
    const v8f vzero = {0.f, 0.f, 0.f, 0.f, 0.f, 0.f, 0.f, 0.f};

    if (layer0) {
        // u = relu((agg0 + x) * W1_row + b1), written in fragment order
        const int kfix = tid & 127;
        const float w1v = W1_0[kfix];
        const float b1v = b1[kfix];
        for (int it = tid; it < 64 * 128; it += 256) {
            int m = it >> 7;
            float v  = agg[row0 + m] + h[row0 + m];
            float uu = v * w1v + b1v;
            sU[frag_idx(m, kfix)] = f2bf(uu > 0.f ? uu : 0.f);
        }
    } else {
        // Preload this wave's 4 A fragments (each element read once per wave)
        const size_t rowoff = (size_t)(row0 + m_base + ncol) * Dd;
        v16bf aF[4];
#pragma unroll
        for (int kc = 0; kc < 4; ++kc)
            aF[kc] = load_frag_A_add(agg, h, rowoff, kc * 32 + half * 8);

#pragma unroll
        for (int nt = 0; nt < 4; ++nt) {
            const int ntg = nt0 + nt;
            v8f acc = vzero;
#pragma unroll
            for (int kc = 0; kc < 4; ++kc)
                acc = wmma_bf16(aF[kc], load_frag_packed(W1f, kc, ntg, lane), acc);
            const float bias = b1[ntg * 16 + ncol];
#pragma unroll
            for (int r = 0; r < 8; ++r) {
                int m = m_base + r + half * 8;
                float uu = acc[r] + bias;
                sU[frag_idx(m, ntg * 16 + ncol)] = f2bf(uu > 0.f ? uu : 0.f);
            }
        }
    }
    __syncthreads();

    // GEMM2: t = u @ W2 + b2, plus per-feature sum / sumsq for BN
    {
        const int m_t = wave & 3;
        v16bf aF[4];
#pragma unroll
        for (int kc = 0; kc < 4; ++kc)
            aF[kc] = load_frag_lds(sU, m_t, kc, lane);

#pragma unroll
        for (int nt = 0; nt < 4; ++nt) {
            const int ntg = nt0 + nt;
            v8f acc = vzero;
#pragma unroll
            for (int kc = 0; kc < 4; ++kc)
                acc = wmma_bf16(aF[kc], load_frag_packed(W2f, kc, ntg, lane), acc);
            const float bias = b2[ntg * 16 + ncol];
            float s = 0.f, sq = 0.f;
#pragma unroll
            for (int r = 0; r < 8; ++r) {
                int m_g = row0 + m_base + r + half * 8;
                float tv = acc[r] + bias;
                t_out[(size_t)m_g * Dd + ntg * 16 + ncol] = tv;
                s  += tv;
                sq += tv * tv;
            }
            atomicAdd(&sums[ntg * 16 + ncol], s);
            atomicAdd(&sums[128 + ntg * 16 + ncol], sq);
        }
    }
}

// ---------------------------------------------------------------------------
// BatchNorm (+optional ReLU) + layerwise global add-pool
// ---------------------------------------------------------------------------
__global__ void bn_pool_kernel(const float* __restrict__ t,
                               const float* __restrict__ sums,
                               const float* __restrict__ gma,
                               const float* __restrict__ bta,
                               const int* __restrict__ batch,
                               float* __restrict__ hout,
                               float* __restrict__ pooled,
                               int layer, int do_relu, int N) {
    int i = blockIdx.x * 256 + threadIdx.x;
    if (i >= N * Dd) return;
    int d = i & 127, n = i >> 7;
    float invN = 1.f / (float)N;
    float mu  = sums[d] * invN;
    float var = sums[128 + d] * invN - mu * mu;
    float v = gma[d] * (t[i] - mu) * rsqrtf(var + BN_EPS) + bta[d];
    if (do_relu) v = v > 0.f ? v : 0.f;
    hout[i] = v;
    atomicAdd(&pooled[(size_t)batch[n] * (Ll * Dd) + layer * Dd + d], v);
}

// ---------------------------------------------------------------------------
// Projection head: z = relu(pooled @ Wp1 + bp1) @ Wp2 + bp2
// pooled [128 x 640] row-major; Wp1 packed as 5 x [128x128] frag blocks.
// One block, 8 waves, wave w owns row tile w.
// ---------------------------------------------------------------------------
__global__ __launch_bounds__(256) void proj_kernel(
    const float* __restrict__ pooled,
    const unsigned short* __restrict__ Wp1f,  // 5 * 16384 u16
    const float* __restrict__ bp1,
    const unsigned short* __restrict__ Wp2f,  // 16384 u16
    const float* __restrict__ bp2,
    float* __restrict__ z) {
    __shared__ unsigned short sU[128 * 128];  // 32 KB fragment order [m_t 8][kc 4][lane][16]

    const int tid    = threadIdx.x;
    const int wave   = tid >> 5;
    const int lane   = tid & 31;
    const int m_base = wave * 16;
    const int half   = (lane >> 4) & 1;
    const int ncol   = lane & 15;
    const v8f vzero = {0.f, 0.f, 0.f, 0.f, 0.f, 0.f, 0.f, 0.f};

    v8f acc[8];
#pragma unroll
    for (int nt = 0; nt < 8; ++nt) acc[nt] = vzero;

    // GEMM1 over K = 640 (20 fragments of 32)
    const size_t rowoff = (size_t)(m_base + ncol) * (Ll * Dd);
    for (int kc = 0; kc < 20; ++kc) {
        v16bf a = load_frag_A_plain(pooled, rowoff, kc * 32 + half * 8);
        const unsigned short* blk = Wp1f + (size_t)(kc >> 2) * 16384;
        const int kcc = kc & 3;
#pragma unroll
        for (int nt = 0; nt < 8; ++nt)
            acc[nt] = wmma_bf16(a, load_frag_packed(blk, kcc, nt, lane), acc[nt]);
    }

    // u = relu(acc + bp1) -> LDS fragment order; sU index uses m_t = m>>4
#pragma unroll
    for (int nt = 0; nt < 8; ++nt) {
        const float bias = bp1[nt * 16 + ncol];
#pragma unroll
        for (int r = 0; r < 8; ++r) {
            int m = m_base + r + half * 8;
            float u = acc[nt][r] + bias;
            int kl = (nt * 16 + ncol) & 31;
            int idx = (((((m >> 4) << 2) + ((nt * 16 + ncol) >> 5)) << 5)
                       + (m & 15) + ((kl >> 3) & 1) * 16) * 16
                      + ((kl >> 4) & 1) * 8 + (kl & 7);
            sU[idx] = f2bf(u > 0.f ? u : 0.f);
        }
    }
    __syncthreads();

    // GEMM2: z = u @ Wp2 + bp2
    v16bf aF[4];
#pragma unroll
    for (int kc = 0; kc < 4; ++kc)
        aF[kc] = load_frag_lds(sU, wave, kc, lane);

#pragma unroll
    for (int nt = 0; nt < 8; ++nt) {
        v8f a2 = vzero;
#pragma unroll
        for (int kc = 0; kc < 4; ++kc)
            a2 = wmma_bf16(aF[kc], load_frag_packed(Wp2f, kc, nt, lane), a2);
        const float bias = bp2[nt * 16 + ncol];
#pragma unroll
        for (int r = 0; r < 8; ++r) {
            int m = m_base + r + half * 8;
            z[(size_t)m * Dd + nt * 16 + ncol] = a2[r] + bias;
        }
    }
}

// ---------------------------------------------------------------------------
// Host launcher
// ---------------------------------------------------------------------------
extern "C" void kernel_launch(void* const* d_in, const int* in_sizes, int n_in,
                              void* d_out, int out_size, void* d_ws, size_t ws_size,
                              hipStream_t stream) {
    const float* x     = (const float*)d_in[0];
    const float* ew    = (const float*)d_in[1];
    const float* W1_0  = (const float*)d_in[2];
    const float* W1_r  = (const float*)d_in[3];
    const float* b1    = (const float*)d_in[4];
    const float* W2    = (const float*)d_in[5];
    const float* b2    = (const float*)d_in[6];
    const float* gamma = (const float*)d_in[7];
    const float* beta  = (const float*)d_in[8];
    const float* Wp1   = (const float*)d_in[9];
    const float* bp1   = (const float*)d_in[10];
    const float* Wp2   = (const float*)d_in[11];
    const float* bp2   = (const float*)d_in[12];
    const int*   eidx  = (const int*)d_in[13];
    const int*   batch = (const int*)d_in[14];
    const int* src = eidx;
    const int* dst = eidx + Ee;

    // Workspace layout
    float* ws     = (float*)d_ws;
    float* hbuf   = ws;                          // N*128 : activations
    float* aggT   = hbuf + (size_t)Nn * Dd;      // N*128 : agg, then t (aliased)
    float* out0   = aggT + (size_t)Nn * Dd;      // N     : layer-0 scalar agg
    float* sums   = out0 + Nn;                   // 256   : BN sum / sumsq
    float* pooled = sums + 256;                  // G * 640
    unsigned short* W1f  = (unsigned short*)(pooled + (size_t)Gg * Ll * Dd);
    unsigned short* W2f  = W1f + 4 * 16384;      // 5 matrices
    unsigned short* Wp1f = W2f + 5 * 16384;      // 5 K-blocks
    unsigned short* Wp2f = Wp1f + 5 * 16384;     // 1 matrix

    // Pre-pack all weights into WMMA fragment order (bf16)
    for (int j = 0; j < 4; ++j)
        pack_weight_kernel<<<64, 256, 0, stream>>>(W1_r + (size_t)j * 16384, W1f + (size_t)j * 16384);
    for (int j = 0; j < 5; ++j)
        pack_weight_kernel<<<64, 256, 0, stream>>>(W2 + (size_t)j * 16384, W2f + (size_t)j * 16384);
    for (int j = 0; j < 5; ++j)
        pack_weight_kernel<<<64, 256, 0, stream>>>(Wp1 + (size_t)j * 16384, Wp1f + (size_t)j * 16384);
    pack_weight_kernel<<<64, 256, 0, stream>>>(Wp2, Wp2f);

    zero_kernel<<<(Gg * Ll * Dd + 255) / 256, 256, 0, stream>>>(pooled, Gg * Ll * Dd);

    for (int i = 0; i < Ll; ++i) {
        zero_kernel<<<1, 256, 0, stream>>>(sums, 256);
        if (i == 0) {
            zero_kernel<<<(Nn + 255) / 256, 256, 0, stream>>>(out0, Nn);
            scatter_d1_kernel<<<(Ee + 255) / 256, 256, 0, stream>>>(x, ew, src, dst, out0, Ee);
            mlp_kernel<<<Nn / 64, 256, 0, stream>>>(out0, x, W1_0, nullptr, b1,
                                                    W2f, b2, aggT, sums, Nn, 1);
        } else {
            zero_kernel<<<(Nn * Dd + 255) / 256, 256, 0, stream>>>(aggT, Nn * Dd);
            scatter_d128_kernel<<<(Ee * 32 + 255) / 256, 256, 0, stream>>>(hbuf, ew, src, dst,
                                                                           aggT, Ee);
            mlp_kernel<<<Nn / 64, 256, 0, stream>>>(aggT, hbuf, nullptr,
                                                    W1f + (size_t)(i - 1) * 16384,
                                                    b1 + i * Dd,
                                                    W2f + (size_t)i * 16384,
                                                    b2 + i * Dd,
                                                    aggT, sums, Nn, 0);
        }
        bn_pool_kernel<<<(Nn * Dd + 255) / 256, 256, 0, stream>>>(
            aggT, sums, gamma + i * Dd, beta + i * Dd, batch,
            hbuf, pooled, i, (i < Ll - 1) ? 1 : 0, Nn);
    }

    proj_kernel<<<1, 256, 0, stream>>>(pooled, Wp1f, bp1, Wp2f, bp2, (float*)d_out);
}
